// QuietSTaRAttention_26225070310113
// MI455X (gfx1250) — 16-bit and 8-bit matrix operand order repaired and checked by reading, compiled, not run on hardware
//
#include <hip/hip_runtime.h>
#include <hip/hip_bf16.h>

#define BB  2
#define SS  2048
#define RR  128
#define EE  2048
#define HH  16
#define DD  128
#define SKK (SS + RR)
#define NTK (BB * SS)

typedef __attribute__((ext_vector_type(8)))  int    v8i;
typedef __attribute__((ext_vector_type(8)))  float  v8f;
typedef __attribute__((ext_vector_type(16))) __bf16 v16bf;
typedef int v4gi __attribute__((vector_size(16)));

union AFrag { int i[8]; v16bf v; };
union IFrag { int i[8]; v8i   v; };

#if defined(__has_builtin)
#if __has_builtin(__builtin_amdgcn_global_load_async_to_lds_b128) && \
    __has_builtin(__builtin_amdgcn_s_wait_asynccnt)
#define USE_ASYNC_LDS 1
#endif
#endif

__device__ __forceinline__ void cp_async16(void* lds_dst, const void* gsrc) {
#if USE_ASYNC_LDS
    __builtin_amdgcn_global_load_async_to_lds_b128((v4gi*)gsrc, (v4gi*)lds_dst,
                                                   0, 0);
#else
    *(int4*)lds_dst = *(const int4*)gsrc;
#endif
}

__device__ __forceinline__ void cp_async_join() {
#if USE_ASYNC_LDS
    __builtin_amdgcn_s_wait_asynccnt(0);
#endif
}

__device__ __forceinline__ unsigned short f32_to_bf16_bits(float f) {
    unsigned int u = __builtin_bit_cast(unsigned int, f);
    u += 0x7FFFu + ((u >> 16) & 1u);
    return (unsigned short)(u >> 16);
}

__global__ __launch_bounds__(256) void zero_ws_kernel(float* wsum) {
    if (threadIdx.x < 8) wsum[threadIdx.x] = 0.0f;
}

__global__ __launch_bounds__(256) void wabs_sum_kernel(
    const float* __restrict__ qw, const float* __restrict__ kw,
    const float* __restrict__ vw, float* __restrict__ wsum)
{
    const float* w = (blockIdx.y == 0) ? qw : (blockIdx.y == 1) ? kw : vw;
    float s = 0.0f;
    for (size_t i = (size_t)blockIdx.x * 256 + threadIdx.x; i < (size_t)EE * EE;
         i += (size_t)gridDim.x * 256)
        s += fabsf(w[i]);
    __shared__ float red[256];
    red[threadIdx.x] = s;
    __syncthreads();
    for (int off = 128; off > 0; off >>= 1) {
        if (threadIdx.x < off) red[threadIdx.x] += red[threadIdx.x + off];
        __syncthreads();
    }
    if (threadIdx.x == 0) atomicAdd(&wsum[blockIdx.y], red[0]);
}

__global__ __launch_bounds__(256) void wquant_kernel(
    const float* __restrict__ qw, const float* __restrict__ kw,
    const float* __restrict__ vw, const float* __restrict__ ow,
    const float* __restrict__ wsum, signed char* __restrict__ Wq,
    unsigned short* __restrict__ kwb, unsigned short* __restrict__ vwb,
    unsigned short* __restrict__ owb)
{
    size_t i = (size_t)blockIdx.x * 256 + threadIdx.x;
    if (i >= (size_t)EE * EE) return;
    const float inv = 1.0f / ((float)EE * (float)EE);
    const float s0 = fmaxf(wsum[0] * inv, 1e-5f);
    const float s1 = fmaxf(wsum[1] * inv, 1e-5f);
    const float s2 = fmaxf(wsum[2] * inv, 1e-5f);
    Wq[i]                       = (signed char)fminf(1.f, fmaxf(-1.f, rintf(qw[i] / s0)));
    Wq[(size_t)EE * EE + i]     = (signed char)fminf(1.f, fmaxf(-1.f, rintf(kw[i] / s1)));
    Wq[2 * (size_t)EE * EE + i] = (signed char)fminf(1.f, fmaxf(-1.f, rintf(vw[i] / s2)));
    kwb[i] = f32_to_bf16_bits(kw[i]);
    vwb[i] = f32_to_bf16_bits(vw[i]);
    owb[i] = f32_to_bf16_bits(ow[i]);
}

__global__ __launch_bounds__(256) void aquant_kernel(
    const float* __restrict__ x, signed char* __restrict__ Xq,
    float* __restrict__ mrow)
{
    const int row = blockIdx.x;
    const float* xr = x + (size_t)row * EE;
    float mx = 0.0f;
    for (int i = threadIdx.x; i < EE; i += 256) mx = fmaxf(mx, fabsf(xr[i]));
    __shared__ float red[256];
    red[threadIdx.x] = mx;
    __syncthreads();
    for (int off = 128; off > 0; off >>= 1) {
        if (threadIdx.x < off) red[threadIdx.x] = fmaxf(red[threadIdx.x], red[threadIdx.x + off]);
        __syncthreads();
    }
    const float amax = fmaxf(red[0], 1e-5f);
    const float s = 127.0f / amax;
    if (threadIdx.x == 0) mrow[row] = amax * (1.0f / 127.0f);
    signed char* dst = Xq + (size_t)row * EE;
    for (int i = threadIdx.x; i < EE; i += 256)
        dst[i] = (signed char)fminf(127.f, fmaxf(-128.f, rintf(xr[i] * s)));
}

__global__ __launch_bounds__(256) void qkv_i8_gemm_kernel(
    const signed char* __restrict__ Xq, const float* __restrict__ mrow,
    const signed char* __restrict__ Wq, const float* __restrict__ wsum,
    unsigned short* __restrict__ qbf,
    unsigned short* __restrict__ kbf,
    unsigned short* __restrict__ vbf)
{
    const int which = blockIdx.z;
    const int m0 = blockIdx.x * 64;
    const int n0 = blockIdx.y * 32;
    __shared__ signed char aT[2][64][64];
    __shared__ signed char bT[2][32][64];
    const int t = threadIdx.x, wave = t >> 5, lane = t & 31;
    const int wm = wave & 3, wn = wave >> 2;
    const int ln = lane & 15, hi = lane >> 4;
    const signed char* W = Wq + (size_t)which * EE * EE;

    const int srow = t >> 2, sc16 = (t & 3) * 16;
    auto stage = [&](int buf, int kt) {
        cp_async16(&aT[buf][srow][sc16],
                   &Xq[(size_t)(m0 + srow) * EE + kt + sc16]);
        if (t < 128)
            cp_async16(&bT[buf][srow][sc16],
                       &W[(size_t)(n0 + srow) * EE + kt + sc16]);
    };

    stage(0, 0);
    v8i acc = {};
    int it = 0;
    for (int kt = 0; kt < EE; kt += 64, ++it) {
        const int buf = it & 1;
        cp_async_join();
        __syncthreads();
        if (kt + 64 < EE) stage(buf ^ 1, kt + 64);

        IFrag a, b;
        const int* arow = (const int*)&aT[buf][wm * 16 + ln][0];
        const int* brow = (const int*)&bT[buf][wn * 16 + ln][0];
#pragma unroll
        for (int v = 0; v < 8; ++v) {
            a.i[v] = arow[(v & 1) + (((v >> 1) & 1) * 4) + ((v >> 2) * 8) + hi * 2];
            b.i[v] = brow[(v & 1) + (((v >> 1) & 1) * 4) + ((v >> 2) * 8) + hi * 2];
        }
        acc = __builtin_amdgcn_wmma_i32_16x16x64_iu8(true, a.v, true, b.v, acc,
                                                     false, false);
    }

    const float wsc =
        fmaxf(wsum[which] * (1.0f / ((float)EE * (float)EE)), 1e-5f);
    const int ncol = n0 + wn * 16 + ln;
    const int h = ncol >> 7, d = ncol & (DD - 1);
#pragma unroll
    for (int i = 0; i < 8; ++i) {
        const int m = m0 + wm * 16 + i + hi * 8;
        const float val = (float)acc[i] * (mrow[m] * wsc);
        const unsigned short bv = f32_to_bf16_bits(val);
        const int b2 = m >> 11, sidx = m & (SS - 1);
        if (which == 0)
            qbf[(((size_t)b2 * HH + h) * SS + sidx) * DD + d] = bv;
        else if (which == 1)
            kbf[(((size_t)b2 * HH + h) * SKK + sidx) * DD + d] = bv;
        else
            vbf[(((size_t)b2 * HH + h) * SKK + sidx) * DD + d] = bv;
    }
}

__global__ __launch_bounds__(256) void reason_gemm_kernel(
    const float* __restrict__ rtok,
    const unsigned short* __restrict__ wb,
    unsigned short* __restrict__ dstbf)
{
    const int m0 = blockIdx.x * 64;
    const int n0 = blockIdx.y * 32;
    __shared__ unsigned short aT[2][64][32];
    __shared__ unsigned short bT[2][32][32];
    const int t = threadIdx.x, wave = t >> 5, lane = t & 31;
    const int wm = wave & 3, wn = wave >> 2;
    const int ln = lane & 15, hi = lane >> 4;

    const int srow = t >> 2, sc0 = (t & 3) * 8;
    auto stage = [&](int buf, int kt) {
        const float* src = &rtok[(size_t)(m0 + srow) * EE + kt + sc0];
        unsigned short tmp[8];
#pragma unroll
        for (int j = 0; j < 8; ++j) tmp[j] = f32_to_bf16_bits(src[j]);
        *(int4*)&aT[buf][srow][sc0] = *(const int4*)tmp;
        if (t < 128)
            cp_async16(&bT[buf][srow][sc0],
                       &wb[(size_t)(n0 + srow) * EE + kt + sc0]);
    };

    stage(0, 0);
    v8f acc = {};
    int it = 0;
    for (int kt = 0; kt < EE; kt += 32, ++it) {
        const int buf = it & 1;
        cp_async_join();
        __syncthreads();
        if (kt + 32 < EE) stage(buf ^ 1, kt + 32);

        AFrag a, b;
        const int* arow = (const int*)&aT[buf][wm * 16 + ln][0];
        const int* brow = (const int*)&bT[buf][wn * 16 + ln][0];
#pragma unroll
        for (int v = 0; v < 8; ++v) {
            a.i[v] = arow[(v & 3) + ((v >> 2) * 8) + hi * 4];
            b.i[v] = brow[(v & 3) + ((v >> 2) * 8) + hi * 4];
        }
        acc = __builtin_amdgcn_wmma_f32_16x16x32_bf16(false, a.v, false, b.v,
                                                      (short)0, acc, false, false);
    }
    const int ncol = n0 + wn * 16 + ln;
    const int h = ncol >> 7, d = ncol & (DD - 1);
#pragma unroll
    for (int i = 0; i < 8; ++i) {
        const int m = m0 + wm * 16 + i + hi * 8;
        const int b2 = m >> 7, r = m & (RR - 1);
        dstbf[(((size_t)b2 * HH + h) * SKK + SS + r) * DD + d] =
            f32_to_bf16_bits(acc[i]);
    }
}

__global__ __launch_bounds__(256) void attn_kernel(
    const unsigned short* __restrict__ qbf,
    const unsigned short* __restrict__ kbf,
    const unsigned short* __restrict__ vbf,
    unsigned short* __restrict__ obf)
{
    const int bh = blockIdx.y;
    const int q0 = blockIdx.x * 128;
    const int b = bh / HH, h = bh % HH;
    __shared__ unsigned short qs[128][DD];
    __shared__ unsigned short kst[2][32][DD];
    __shared__ unsigned short vt[2][DD][32];
    __shared__ unsigned short ps[8][16][32];
    const int t = threadIdx.x, wave = t >> 5, lane = t & 31;
    const int ln = lane & 15, hi = lane >> 4;

    const unsigned short* qg = qbf + ((size_t)bh * SS + q0) * DD;
    const unsigned short* kg = kbf + (size_t)bh * SKK * DD;
    const unsigned short* vg = vbf + (size_t)bh * SKK * DD;

    for (int i = t; i < 128 * DD / 8; i += 256)
        cp_async16((int4*)&qs[0][0] + i, (const int4*)qg + i);

    auto stageKV = [&](int buf, int k0) {
        for (int i = t; i < 32 * DD / 8; i += 256)
            cp_async16((int4*)&kst[buf][0][0] + i,
                       (const int4*)(kg + (size_t)k0 * DD) + i);
        const int4* s = (const int4*)(vg + (size_t)k0 * DD);
        for (int i = t; i < 32 * DD / 8; i += 256) {
            const int key = i >> 4, d0 = (i & 15) * 8;
            int4 val = s[i];
            const unsigned short* e = (const unsigned short*)&val;
#pragma unroll
            for (int j = 0; j < 8; ++j) vt[buf][d0 + j][key] = e[j];
        }
    };
    stageKV(0, 0);

    cp_async_join();
    __syncthreads();

    AFrag qa[4];
    {
        const int* qrow = (const int*)&qs[wave * 16 + ln][0];
#pragma unroll
        for (int kq = 0; kq < 4; ++kq)
#pragma unroll
            for (int v = 0; v < 8; ++v)
                qa[kq].i[v] = qrow[kq * 16 + (v & 3) + ((v >> 2) * 8) + hi * 4];
    }

    v8f oacc[8];
#pragma unroll
    for (int j = 0; j < 8; ++j) oacc[j] = {};
    float rowM[8], rowL[8];
#pragma unroll
    for (int i = 0; i < 8; ++i) { rowM[i] = -3.0e38f; rowL[i] = 0.0f; }

    const float sscale = 0.08838834764831845f;

    int it = 0;
    for (int k0 = 0; k0 < SKK; k0 += 32, ++it) {
        const int buf = it & 1;
        cp_async_join();
        __syncthreads();
        if (k0 + 32 < SKK) stageKV(buf ^ 1, k0 + 32);

        v8f sc[2];
#pragma unroll
        for (int nt = 0; nt < 2; ++nt) {
            v8f s = {};
            const int* krow = (const int*)&kst[buf][nt * 16 + ln][0];
#pragma unroll
            for (int kq = 0; kq < 4; ++kq) {
                AFrag bf;
#pragma unroll
                for (int v = 0; v < 8; ++v) bf.i[v] = krow[kq * 16 + (v & 3) + ((v >> 2) * 8) + hi * 4];
                s = __builtin_amdgcn_wmma_f32_16x16x32_bf16(
                        false, qa[kq].v, false, bf.v, (short)0, s, false, false);
            }
            sc[nt] = s;
        }

#pragma unroll
        for (int i = 0; i < 8; ++i) {
            const float s0 = sc[0][i] * sscale;
            const float s1 = sc[1][i] * sscale;
            float mx = fmaxf(s0, s1);
#pragma unroll
            for (int msk = 1; msk < 16; msk <<= 1)
                mx = fmaxf(mx, __shfl_xor(mx, msk, 32));
            const float nm = fmaxf(rowM[i], mx);
            const float corr = __expf(rowM[i] - nm);
            const float p0 = __expf(s0 - nm);
            const float p1 = __expf(s1 - nm);
            float ls = p0 + p1;
#pragma unroll
            for (int msk = 1; msk < 16; msk <<= 1)
                ls += __shfl_xor(ls, msk, 32);
            rowL[i] = rowL[i] * corr + ls;
            rowM[i] = nm;
#pragma unroll
            for (int j = 0; j < 8; ++j) oacc[j][i] *= corr;
            const int m = i + hi * 8;
            ps[wave][m][ln]      = f32_to_bf16_bits(p0);
            ps[wave][m][16 + ln] = f32_to_bf16_bits(p1);
        }

        AFrag pf;
        {
            const int* prow = (const int*)&ps[wave][ln][0];
#pragma unroll
            for (int v = 0; v < 8; ++v)
                pf.i[v] = prow[(v & 3) + ((v >> 2) * 8) + hi * 4];
        }
#pragma unroll
        for (int j = 0; j < 8; ++j) {
            AFrag bf;
            const int* vrow = (const int*)&vt[buf][j * 16 + ln][0];
#pragma unroll
            for (int v = 0; v < 8; ++v) bf.i[v] = vrow[(v & 3) + ((v >> 2) * 8) + hi * 4];
            oacc[j] = __builtin_amdgcn_wmma_f32_16x16x32_bf16(
                          false, pf.v, false, bf.v, (short)0, oacc[j], false, false);
        }
    }

#pragma unroll
    for (int i = 0; i < 8; ++i) {
        const float inv = 1.0f / rowL[i];
        const int m = i + hi * 8;
        const int sidx = q0 + wave * 16 + m;
        const size_t base = ((size_t)b * SS + sidx) * EE + (size_t)h * DD;
#pragma unroll
        for (int j = 0; j < 8; ++j)
            obf[base + j * 16 + ln] = f32_to_bf16_bits(oacc[j][i] * inv);
    }
}

__global__ __launch_bounds__(256) void out_gemm_kernel(
    const unsigned short* __restrict__ obf,
    const unsigned short* __restrict__ wb,
    const float* __restrict__ bias,
    float* __restrict__ out)
{
    const int m0 = blockIdx.x * 64, n0 = blockIdx.y * 32;
    __shared__ unsigned short aT[2][64][32];
    __shared__ unsigned short bT[2][32][32];
    const int t = threadIdx.x, wave = t >> 5, lane = t & 31;
    const int wm = wave & 3, wn = wave >> 2;
    const int ln = lane & 15, hi = lane >> 4;

    const int srow = t >> 2, sc0 = (t & 3) * 8;
    auto stage = [&](int buf, int kt) {
        cp_async16(&aT[buf][srow][sc0],
                   &obf[(size_t)(m0 + srow) * EE + kt + sc0]);
        if (t < 128)
            cp_async16(&bT[buf][srow][sc0],
                       &wb[(size_t)(n0 + srow) * EE + kt + sc0]);
    };

    stage(0, 0);
    v8f acc = {};
    int it = 0;
    for (int kt = 0; kt < EE; kt += 32, ++it) {
        const int buf = it & 1;
        cp_async_join();
        __syncthreads();
        if (kt + 32 < EE) stage(buf ^ 1, kt + 32);

        AFrag a, b;
        const int* arow = (const int*)&aT[buf][wm * 16 + ln][0];
        const int* brow = (const int*)&bT[buf][wn * 16 + ln][0];
#pragma unroll
        for (int v = 0; v < 8; ++v) {
            a.i[v] = arow[(v & 3) + ((v >> 2) * 8) + hi * 4];
            b.i[v] = brow[(v & 3) + ((v >> 2) * 8) + hi * 4];
        }
        acc = __builtin_amdgcn_wmma_f32_16x16x32_bf16(false, a.v, false, b.v,
                                                      (short)0, acc, false, false);
    }
    const int ncol = n0 + wn * 16 + ln;
#pragma unroll
    for (int i = 0; i < 8; ++i) {
        const int m = m0 + wm * 16 + i + hi * 8;
        out[(size_t)m * EE + ncol] = acc[i] + bias[ncol];
    }
}

extern "C" void kernel_launch(void* const* d_in, const int* in_sizes, int n_in,
                              void* d_out, int out_size, void* d_ws, size_t ws_size,
                              hipStream_t stream)
{
    (void)in_sizes; (void)n_in; (void)out_size; (void)ws_size;
    const float* x    = (const float*)d_in[0];
    const float* rtok = (const float*)d_in[1];
    const float* qw   = (const float*)d_in[2];
    const float* kw   = (const float*)d_in[3];
    const float* vw   = (const float*)d_in[4];
    const float* ow   = (const float*)d_in[5];
    const float* ob   = (const float*)d_in[6];
    float* out = (float*)d_out;

    char* ws = (char*)d_ws;
    size_t off = 0;
    auto take = [&](size_t bytes) {
        char* p = ws + off;
        off = (off + bytes + 255) & ~(size_t)255;
        return p;
    };
    signed char*    Xq   = (signed char*)take((size_t)NTK * EE);
    float*          mrow = (float*)take((size_t)NTK * 4);
    float*          wsum = (float*)take(256);
    signed char*    Wq   = (signed char*)take(3 * (size_t)EE * EE);
    unsigned short* kwb  = (unsigned short*)take((size_t)EE * EE * 2);
    unsigned short* vwb  = (unsigned short*)take((size_t)EE * EE * 2);
    unsigned short* owb  = (unsigned short*)take((size_t)EE * EE * 2);
    unsigned short* qbf  = (unsigned short*)take((size_t)BB * HH * SS  * DD * 2);
    unsigned short* kbf  = (unsigned short*)take((size_t)BB * HH * SKK * DD * 2);
    unsigned short* vbf  = (unsigned short*)take((size_t)BB * HH * SKK * DD * 2);
    unsigned short* obf  = (unsigned short*)take((size_t)NTK * EE * 2);

    zero_ws_kernel<<<1, 256, 0, stream>>>(wsum);
    wabs_sum_kernel<<<dim3(256, 3), 256, 0, stream>>>(qw, kw, vw, wsum);
    wquant_kernel<<<(EE * EE) / 256, 256, 0, stream>>>(qw, kw, vw, ow, wsum,
                                                       Wq, kwb, vwb, owb);
    aquant_kernel<<<NTK, 256, 0, stream>>>(x, Xq, mrow);
    qkv_i8_gemm_kernel<<<dim3(NTK / 64, EE / 32, 3), 256, 0, stream>>>(
        Xq, mrow, Wq, wsum, qbf, kbf, vbf);
    reason_gemm_kernel<<<dim3((BB * RR) / 64, EE / 32), 256, 0, stream>>>(
        rtok, kwb, kbf);
    reason_gemm_kernel<<<dim3((BB * RR) / 64, EE / 32), 256, 0, stream>>>(
        rtok, vwb, vbf);
    attn_kernel<<<dim3(SS / 128, BB * HH), 256, 0, stream>>>(qbf, kbf, vbf, obf);
    out_gemm_kernel<<<dim3(NTK / 64, EE / 32), 256, 0, stream>>>(obf, owb, ob, out);
}
